// Rec_WindowAttention_31705448579795
// MI455X (gfx1250) — matrix operand order repaired and checked by reading, not compiled, not run on hardware
//
#include <hip/hip_runtime.h>

typedef _Float16 h16;
typedef __attribute__((ext_vector_type(16))) _Float16 v16h;
typedef __attribute__((ext_vector_type(8)))  _Float16 v8h;
typedef __attribute__((ext_vector_type(8)))  float    v8f;

union Frag { v16h v; v8h h[2]; };

// ---------------- LDS layout (dynamic shared), all offsets 16B aligned ----------------
#define STR   264           // row stride (halves) for [64][C] f16 buffers (padded vs 256)
#define VSTRD 72            // row stride (halves) for vT [256][64] f16 buffer (padded)
#define PSTR  72            // row stride (halves) for per-wave P scratch [16][64]
#define XT_OFF   0          // fp32 x tile [64][256]           (65536 B)  -- phase 1/2 only
#define PS_OFF   0          // P scratch, 8 waves * 16*PSTR*2  (18432 B)  -- reuses XT
#define OB_OFF   20480      // attn output f16 [64][STR]       (33792 B)  -- reuses XT
#define DWQ_OFF  65536
#define DWK_OFF  99328
#define DWV_OFF  133120
#define QF_OFF   166912
#define KF_OFF   200704
#define VT_OFF   234496     // 256*VSTRD*2 = 36864 B
#define SMEM_BYTES 271360

__device__ __forceinline__ v8f zero8() {
  v8f z = {0.f,0.f,0.f,0.f,0.f,0.f,0.f,0.f};
  return z;
}
__device__ __forceinline__ v8f wmma16(v16h a, v16h b, v8f c) {
  return __builtin_amdgcn_wmma_f32_16x16x32_f16(false, a, false, b, (short)0, c, false, false);
}

// ---------------- prologue: fp32->f16 weight conversion + dense bias gather ----------------
__global__ __launch_bounds__(256) void prep_kernel(
    const float* __restrict__ qpw, const float* __restrict__ kpw,
    const float* __restrict__ vpw, const float* __restrict__ pjw,
    const float* __restrict__ bias_table, const int* __restrict__ rel_index,
    h16* __restrict__ w16, float* __restrict__ biasAll)
{
  int idx = blockIdx.x * 256 + threadIdx.x;
  if (idx < 4 * 65536) {
    int m = idx >> 16, r = idx & 65535;
    const float* src = (m == 0) ? qpw : (m == 1) ? kpw : (m == 2) ? vpw : pjw;
    w16[idx] = (h16)src[r];
  } else {
    int t = idx - 4 * 65536;            // t = h*4096 + i*64 + j
    if (t < 8 * 4096) {
      int h = t >> 12, ij = t & 4095;
      biasAll[t] = bias_table[rel_index[ij] * 8 + h];
    }
  }
}

// 64x256 = [64x256 f16 LDS] x [256x256 f16 global]^T per documented wave32 WMMA
// 16-bit A/B layouts. Each wave owns 2 N-tiles; per k-step: preload the 4 A-frags
// (one ds clause, one wait), then issue 4 back-to-back v_wmma.
template<bool TRANSPOSED, int DSTR>
__device__ __forceinline__ void pw_gemm(const h16* __restrict__ Asrc,
                                        const h16* __restrict__ Bw,
                                        const float* __restrict__ bias,
                                        h16* __restrict__ Dst,
                                        int wave, int L, int hi)
{
#pragma unroll
  for (int nn = 0; nn < 2; ++nn) {
    const int nt = wave * 2 + nn;
    const int co = nt * 16 + L;
    const float bv = bias[co];
    v8f acc[4];
#pragma unroll
    for (int i = 0; i < 4; ++i) acc[i] = zero8();
#pragma unroll
    for (int ks = 0; ks < 8; ++ks) {
      Frag bf;                                        // B: lane = col co, K striped
      const h16* bp = Bw + co * 256 + ks * 32 + hi * 8;
      bf.h[0] = *(const v8h*)(bp);
      bf.h[1] = *(const v8h*)(bp + 16);
      Frag af[4];                                     // A: preload all 4 M-tiles
#pragma unroll
      for (int mt = 0; mt < 4; ++mt) {
        const h16* ap = Asrc + (mt * 16 + L) * STR + ks * 32 + hi * 8;
        af[mt].h[0] = *(const v8h*)(ap);
        af[mt].h[1] = *(const v8h*)(ap + 16);
      }
#pragma unroll
      for (int mt = 0; mt < 4; ++mt)
        acc[mt] = wmma16(af[mt].v, bf.v, acc[mt]);
    }
#pragma unroll
    for (int mt = 0; mt < 4; ++mt) {
#pragma unroll
      for (int g = 0; g < 8; ++g) {                   // C layout: VGPR g -> row g / g+8
        float val = acc[mt][g] + bv;
        int m = mt * 16 + g + hi * 8;
        if (TRANSPOSED) Dst[co * DSTR + m] = (h16)val;
        else            Dst[m * DSTR + co] = (h16)val;
      }
    }
  }
}

__global__ __launch_bounds__(256, 1) void win_attn_main(
    const float* __restrict__ x,
    const float* __restrict__ q_dw_w, const float* __restrict__ q_dw_b,
    const float* __restrict__ k_dw_w, const float* __restrict__ k_dw_b,
    const float* __restrict__ v_dw_w, const float* __restrict__ v_dw_b,
    const float* __restrict__ q_pw_b, const float* __restrict__ k_pw_b,
    const float* __restrict__ v_pw_b,
    const h16* __restrict__ wq16, const h16* __restrict__ wk16,
    const h16* __restrict__ wv16, const h16* __restrict__ wp16,
    const float* __restrict__ biasAll, const float* __restrict__ proj_b,
    float* __restrict__ out)
{
  extern __shared__ char smem[];
  const int tid  = threadIdx.x;
  const int wave = tid >> 5;
  const int lane = tid & 31;
  const int L    = lane & 15;
  const int hi   = lane >> 4;
  const int b    = blockIdx.x;

  float* xt  = (float*)(smem + XT_OFF);
  h16* dwq   = (h16*)(smem + DWQ_OFF);
  h16* dwk   = (h16*)(smem + DWK_OFF);
  h16* dwv   = (h16*)(smem + DWV_OFF);
  h16* qf    = (h16*)(smem + QF_OFF);
  h16* kf    = (h16*)(smem + KF_OFF);
  h16* vT    = (h16*)(smem + VT_OFF);

  // ---- phase 1: stage x[b] (64x256 fp32) into LDS, coalesced b128 loads ----
  {
    const float4* xg = (const float4*)(x + (size_t)b * 64 * 256);
    float4* xs = (float4*)xt;
#pragma unroll
    for (int i = 0; i < 16; ++i) xs[tid + i * 256] = xg[tid + i * 256];
  }
  __syncthreads();

  // ---- phase 2: depthwise 3x3 + bias + ReLU for q/k/v, thread = channel ----
  {
    const int c = tid;
    float wgt[27];
#pragma unroll
    for (int t = 0; t < 9; ++t) {
      wgt[t]      = q_dw_w[c * 9 + t];
      wgt[9 + t]  = k_dw_w[c * 9 + t];
      wgt[18 + t] = v_dw_w[c * 9 + t];
    }
    const float bq = q_dw_b[c], bk = k_dw_b[c], bv = v_dw_b[c];
    for (int n = 0; n < 64; ++n) {
      const int i = n >> 3, j = n & 7;
      float aq = bq, ak = bk, av = bv;
#pragma unroll
      for (int di = 0; di < 3; ++di) {
        const int ii = i + di - 1;
        if ((unsigned)ii < 8u) {
#pragma unroll
          for (int dj = 0; dj < 3; ++dj) {
            const int jj = j + dj - 1;
            if ((unsigned)jj < 8u) {
              const float xv = xt[(ii * 8 + jj) * 256 + c];
              const int wI = di * 3 + dj;
              aq += wgt[wI] * xv; ak += wgt[9 + wI] * xv; av += wgt[18 + wI] * xv;
            }
          }
        }
      }
      dwq[n * STR + c] = (h16)fmaxf(aq, 0.f);
      dwk[n * STR + c] = (h16)fmaxf(ak, 0.f);
      dwv[n * STR + c] = (h16)fmaxf(av, 0.f);
    }
  }
  __syncthreads();

  // ---- phase 3: three pointwise GEMMs (WMMA). V output stored transposed. ----
  pw_gemm<false, STR>  (dwq, wq16, q_pw_b, qf, wave, L, hi);
  pw_gemm<false, STR>  (dwk, wk16, k_pw_b, kf, wave, L, hi);
  pw_gemm<true,  VSTRD>(dwv, wv16, v_pw_b, vT, wave, L, hi);
  __syncthreads();

  // ---- phase 4: attention, one wave per head ----
  {
    const int hH = wave;
    const float scale = 0.1767766952966369f;        // 32^-0.5
    h16* ps = (h16*)(smem + PS_OFF) + hH * 16 * PSTR;
    h16* ob = (h16*)(smem + OB_OFF);
    const float* bh = biasAll + hH * 64 * 64;
#pragma unroll 1
    for (int mt = 0; mt < 4; ++mt) {
      Frag af;                                       // q rows for this mtile (K=dh=32)
      const h16* ap = qf + (mt * 16 + L) * STR + hH * 32 + hi * 8;
      af.h[0] = *(const v8h*)ap; af.h[1] = *(const v8h*)(ap + 16);
      Frag bf[4];                                    // preload all 4 k-tiles
#pragma unroll
      for (int nt = 0; nt < 4; ++nt) {
        const h16* bp = kf + (nt * 16 + L) * STR + hH * 32 + hi * 8;
        bf[nt].h[0] = *(const v8h*)bp; bf[nt].h[1] = *(const v8h*)(bp + 16);
      }
      v8f s[4];
#pragma unroll
      for (int nt = 0; nt < 4; ++nt)                 // S = q . k^T, single K step
        s[nt] = wmma16(af.v, bf[nt].v, zero8());
#pragma unroll
      for (int nt = 0; nt < 4; ++nt)
#pragma unroll
        for (int g = 0; g < 8; ++g) {
          const int m = mt * 16 + g + hi * 8;
          s[nt][g] = s[nt][g] * scale + bh[m * 64 + nt * 16 + L];
        }
      // softmax over n: 4 tiles per lane + 16-lane xor reduction (rows stay in group)
#pragma unroll
      for (int g = 0; g < 8; ++g) {
        float mx = fmaxf(fmaxf(s[0][g], s[1][g]), fmaxf(s[2][g], s[3][g]));
        mx = fmaxf(mx, __shfl_xor(mx, 1, 32));
        mx = fmaxf(mx, __shfl_xor(mx, 2, 32));
        mx = fmaxf(mx, __shfl_xor(mx, 4, 32));
        mx = fmaxf(mx, __shfl_xor(mx, 8, 32));
        float sum = 0.f;
#pragma unroll
        for (int nt = 0; nt < 4; ++nt) { float p = __expf(s[nt][g] - mx); s[nt][g] = p; sum += p; }
        sum += __shfl_xor(sum, 1, 32);
        sum += __shfl_xor(sum, 2, 32);
        sum += __shfl_xor(sum, 4, 32);
        sum += __shfl_xor(sum, 8, 32);
        const float r = 1.f / sum;
#pragma unroll
        for (int nt = 0; nt < 4; ++nt) s[nt][g] *= r;
      }
      // P -> f16 wave-private scratch (re-layout C-layout -> A-layout via LDS)
#pragma unroll
      for (int nt = 0; nt < 4; ++nt)
#pragma unroll
        for (int g = 0; g < 8; ++g)
          ps[(g + hi * 8) * PSTR + nt * 16 + L] = (h16)s[nt][g];
      // O = P @ V  (K=64 -> 2 steps; N=dh=32 -> 2 tiles; B from transposed vT)
      v8f o[2]; o[0] = zero8(); o[1] = zero8();
#pragma unroll
      for (int ks = 0; ks < 2; ++ks) {
        Frag pa;
        const h16* pp = ps + L * PSTR + ks * 32 + hi * 8;
        pa.h[0] = *(const v8h*)pp; pa.h[1] = *(const v8h*)(pp + 16);
        Frag vb[2];
#pragma unroll
        for (int nt2 = 0; nt2 < 2; ++nt2) {
          const h16* vp = vT + (hH * 32 + nt2 * 16 + L) * VSTRD + ks * 32 + hi * 8;
          vb[nt2].h[0] = *(const v8h*)vp; vb[nt2].h[1] = *(const v8h*)(vp + 16);
        }
#pragma unroll
        for (int nt2 = 0; nt2 < 2; ++nt2)
          o[nt2] = wmma16(pa.v, vb[nt2].v, o[nt2]);
      }
#pragma unroll
      for (int nt2 = 0; nt2 < 2; ++nt2)
#pragma unroll
        for (int g = 0; g < 8; ++g)
          ob[(mt * 16 + g + hi * 8) * STR + hH * 32 + nt2 * 16 + L] = (h16)o[nt2][g];
    }
  }
  __syncthreads();

  // ---- phase 5: output projection GEMM + bias, fp32 stores to HBM ----
  {
    const h16* ob = (const h16*)(smem + OB_OFF);
    float* outp = out + (size_t)b * 64 * 256;
#pragma unroll
    for (int nn = 0; nn < 2; ++nn) {
      const int nt = wave * 2 + nn;
      const int co = nt * 16 + L;
      const float bv = proj_b[co];
      v8f acc[4];
#pragma unroll
      for (int i = 0; i < 4; ++i) acc[i] = zero8();
#pragma unroll
      for (int ks = 0; ks < 8; ++ks) {
        Frag bf;
        const h16* bp = wp16 + co * 256 + ks * 32 + hi * 8;
        bf.h[0] = *(const v8h*)(bp);
        bf.h[1] = *(const v8h*)(bp + 16);
        Frag af[4];
#pragma unroll
        for (int mt = 0; mt < 4; ++mt) {
          const h16* ap = ob + (mt * 16 + L) * STR + ks * 32 + hi * 8;
          af[mt].h[0] = *(const v8h*)(ap);
          af[mt].h[1] = *(const v8h*)(ap + 16);
        }
#pragma unroll
        for (int mt = 0; mt < 4; ++mt)
          acc[mt] = wmma16(af[mt].v, bf.v, acc[mt]);
      }
#pragma unroll
      for (int mt = 0; mt < 4; ++mt)
#pragma unroll
        for (int g = 0; g < 8; ++g) {
          const int m = mt * 16 + g + hi * 8;
          outp[m * 256 + co] = acc[mt][g] + bv;
        }
    }
  }
}

extern "C" void kernel_launch(void* const* d_in, const int* in_sizes, int n_in,
                              void* d_out, int out_size, void* d_ws, size_t ws_size,
                              hipStream_t stream) {
  (void)in_sizes; (void)n_in; (void)out_size; (void)ws_size;
  const float* x      = (const float*)d_in[0];
  const float* q_dw_w = (const float*)d_in[1];
  const float* q_dw_b = (const float*)d_in[2];
  const float* q_pw_w = (const float*)d_in[3];
  const float* q_pw_b = (const float*)d_in[4];
  const float* k_dw_w = (const float*)d_in[5];
  const float* k_dw_b = (const float*)d_in[6];
  const float* k_pw_w = (const float*)d_in[7];
  const float* k_pw_b = (const float*)d_in[8];
  const float* v_dw_w = (const float*)d_in[9];
  const float* v_dw_b = (const float*)d_in[10];
  const float* v_pw_w = (const float*)d_in[11];
  const float* v_pw_b = (const float*)d_in[12];
  const float* bias_table = (const float*)d_in[13];
  const float* proj_w = (const float*)d_in[14];
  const float* proj_b = (const float*)d_in[15];
  const int*   rel_index = (const int*)d_in[16];

  h16*   w16     = (h16*)d_ws;                                   // 4 x 256x256 f16
  float* biasAll = (float*)((char*)d_ws + (size_t)4 * 65536 * sizeof(h16));

  prep_kernel<<<1152, 256, 0, stream>>>(q_pw_w, k_pw_w, v_pw_w, proj_w,
                                        bias_table, rel_index, w16, biasAll);

  (void)hipFuncSetAttribute(reinterpret_cast<const void*>(win_attn_main),
                            hipFuncAttributeMaxDynamicSharedMemorySize, SMEM_BYTES);

  win_attn_main<<<4096, 256, SMEM_BYTES, stream>>>(
      x, q_dw_w, q_dw_b, k_dw_w, k_dw_b, v_dw_w, v_dw_b,
      q_pw_b, k_pw_b, v_pw_b,
      w16, w16 + 65536, w16 + 2 * 65536, w16 + 3 * 65536,
      biasAll, proj_b, (float*)d_out);
}
